// GNBlock_16733192585484
// MI455X (gfx1250) — matrix operand order repaired and checked by reading, not compiled, not run on hardware
//
#include <hip/hip_runtime.h>
#include <hip/hip_bf16.h>
#include <cstdint>
#include <cstddef>

#define D 128
#define NN 50000
#define NE 800000
#define WV 2    // waves per block (edge kernel)
#define MT 4    // M-tiles (16 rows each) per wave in edge kernel -> 64 edges/wave
#define NWV 4   // waves per block (node kernel)

typedef __attribute__((ext_vector_type(16))) __bf16 v16bf;
typedef __attribute__((ext_vector_type(8)))  __bf16 v8bf;
typedef __attribute__((ext_vector_type(8)))  float  v8f;
typedef uint32_t u32x4 __attribute__((ext_vector_type(4)));
typedef uint32_t u32x8 __attribute__((ext_vector_type(8)));

#define WMMA_BF16(A, B, C) \
    __builtin_amdgcn_wmma_f32_16x16x32_bf16(false, (A), false, (B), (short)0, (C), false, false)

__device__ __forceinline__ float silu_f(float v) {
    return v * (1.0f / (1.0f + __expf(-v)));
}

// A-fragment (16x32 bf16). ISA 7.12.2: lanes 0-15 hold row M=lane, K={0..7} in V0-3 and
// {16..23} in V4-7; lanes 16-31 hold K={8..15} and {24..31}. k2 = kb + (lane>>4)*8.
__device__ __forceinline__ v16bf load_fragA_bf16(const __bf16* row_base, int k2) {
    v8bf lo = *(const v8bf*)(row_base + k2);
    v8bf hi = *(const v8bf*)(row_base + k2 + 16);
    return __builtin_shufflevector(lo, hi, 0,1,2,3,4,5,6,7,8,9,10,11,12,13,14,15);
}

// Same A layout, source f32 converted on the fly (optionally scaled).
__device__ __forceinline__ v16bf cvt_fragA_f32(const float* p, float scale) {
    float4 f0 = ((const float4*)p)[0];
    float4 f1 = ((const float4*)p)[1];
    float4 f2 = ((const float4*)(p + 16))[0];
    float4 f3 = ((const float4*)(p + 16))[1];
    v16bf a;
    a[0]=(__bf16)(f0.x*scale); a[1]=(__bf16)(f0.y*scale); a[2]=(__bf16)(f0.z*scale); a[3]=(__bf16)(f0.w*scale);
    a[4]=(__bf16)(f1.x*scale); a[5]=(__bf16)(f1.y*scale); a[6]=(__bf16)(f1.z*scale); a[7]=(__bf16)(f1.w*scale);
    a[8]=(__bf16)(f2.x*scale); a[9]=(__bf16)(f2.y*scale); a[10]=(__bf16)(f2.z*scale); a[11]=(__bf16)(f2.w*scale);
    a[12]=(__bf16)(f3.x*scale); a[13]=(__bf16)(f3.y*scale); a[14]=(__bf16)(f3.z*scale); a[15]=(__bf16)(f3.w*scale);
    return a;
}

// B-fragment (32x16 bf16) from transposed weights Wt[n][k], same K order as A:
// lanes 0-15: N=lane, K={kb..kb+7} and {kb+16..kb+23}; lanes 16-31: N=lane-16, K={kb+8..kb+15} and {kb+24..kb+31}.
// k_lane = kb + (lane>>4)*8.
__device__ __forceinline__ v16bf load_fragB_bf16(const __bf16* wt, int stride, int col, int k_lane) {
    const __bf16* p = wt + col * stride + k_lane;
    v8bf lo = *(const v8bf*)(p);
    v8bf hi = *(const v8bf*)(p + 16);
    return __builtin_shufflevector(lo, hi, 0,1,2,3,4,5,6,7,8,9,10,11,12,13,14,15);
}

// 1-D Tensor-Data-Mover load: nelem elements of 4 bytes, global -> LDS.
// D# packing per CDNA5 ISA (group0 128b, group1 256b).
__device__ __forceinline__ void tdm_load_1d_f32(uint32_t lds_off, const void* gptr, uint32_t nelem) {
    uint64_t ga = (uint64_t)(uintptr_t)gptr;
    u32x4 g0;
    g0[0] = 1u;                                                   // count=1 (valid user D#)
    g0[1] = lds_off;                                              // lds_addr (bytes)
    g0[2] = (uint32_t)ga;                                         // global_addr[31:0]
    g0[3] = ((uint32_t)(ga >> 32) & 0x01FFFFFFu) | (2u << 30);    // global_addr[56:32] | type=2
    u32x8 g1;
    g1[0] = 2u << 16;                                             // data_size=2 (4B), mask/flags=0
    g1[1] = (nelem & 0xFFFFu) << 16;                              // tensor_dim0[15:0]   (bits 63:48)
    g1[2] = ((nelem >> 16) & 0xFFFFu) | (1u << 16);               // tensor_dim0[31:16], tensor_dim1=1
    g1[3] = (nelem & 0xFFFFu) << 16;                              // tile_dim0           (bits 127:112)
    g1[4] = 0u;                                                   // tile_dim1=0, tile_dim2=0
    g1[5] = nelem;                                                // tensor_dim0_stride[31:0]
    g1[6] = 0u;
    g1[7] = 0u;
    asm volatile("tensor_load_to_lds %0, %1" :: "s"(g0), "s"(g1) : "memory");
}

// ---------------------------------------------------------------------------
// Kernel 0: zero accumulators, convert+transpose weights to bf16 Wt[n][k]
// ---------------------------------------------------------------------------
__global__ void init_kernel(const float* __restrict__ We1, const float* __restrict__ We2,
                            const float* __restrict__ Wn1, const float* __restrict__ Wn2,
                            float* __restrict__ sums, float* __restrict__ cnt,
                            __bf16* __restrict__ wte1, __bf16* __restrict__ wte2,
                            __bf16* __restrict__ wtn1, __bf16* __restrict__ wtn2) {
    const int tid = blockIdx.x * blockDim.x + threadIdx.x;
    const int stp = gridDim.x * blockDim.x;
    for (int i = tid; i < NN * D; i += stp) sums[i] = 0.0f;
    for (int i = tid; i < NN; i += stp) cnt[i] = 0.0f;
    for (int i = tid; i < 3 * D * D; i += stp) { int k = i / D, n = i % D; wte1[n * (3 * D) + k] = (__bf16)We1[i]; }
    for (int i = tid; i < D * D; i += stp)     { int k = i / D, n = i % D; wte2[n * D + k]       = (__bf16)We2[i]; }
    for (int i = tid; i < 2 * D * D; i += stp) { int k = i / D, n = i % D; wtn1[n * (2 * D) + k] = (__bf16)Wn1[i]; }
    for (int i = tid; i < D * D; i += stp)     { int k = i / D, n = i % D; wtn2[n * D + k]       = (__bf16)Wn2[i]; }
}

// ---------------------------------------------------------------------------
// Kernel 1: edge MLP + scatter-add. One wave = 4 M-tiles (64 edges); B fragments
// are reused across the 4 tiles to cut weight (L2) traffic 4x.
// ---------------------------------------------------------------------------
__global__ __launch_bounds__(32 * WV) void edge_kernel(
    const float* __restrict__ x, const int* __restrict__ ei, const float* __restrict__ ea,
    const __bf16* __restrict__ wte1, const float* __restrict__ be1,
    const __bf16* __restrict__ wte2, const float* __restrict__ be2,
    float* __restrict__ out_e, float* __restrict__ sums, float* __restrict__ cnt) {
    __shared__ float  sbias[2 * D];             // be1 | be2, staged via TDM
    __shared__ __bf16 sh_h[WV][MT * 16 * 136];  // h slab: 64 rows x 128 (pad 136)
    __shared__ float  sh_o[WV][16 * 132];       // per-M-tile f32 output staging

    const int wave = threadIdx.x >> 5;
    const int lane = threadIdx.x & 31;

    // ---- stage biases into LDS with the Tensor Data Mover (wave 0 only) ----
    if (wave == 0) {
        const uint32_t bias_lds = (uint32_t)(uintptr_t)(void*)sbias;  // low 32b = LDS offset
        tdm_load_1d_f32(bias_lds, be1, D);
        tdm_load_1d_f32(bias_lds + D * sizeof(float), be2, D);
        __builtin_amdgcn_s_wait_tensorcnt(0);
    }
    __syncthreads();

    const int tile0 = (blockIdx.x * WV + wave) * MT;  // exact: NE/16 tiles total
    const int r = lane & 15;
    const int hf = lane >> 4;
    const int kA = hf * 8;
    const int kB = hf * 8;

    // per-M-tile concat segment pointers (edge_attr | x[row] | x[col])
    const float* segp[MT][3];
    #pragma unroll
    for (int mi = 0; mi < MT; mi++) {
        const int e_r = (tile0 + mi) * 16 + r;
        const int ridx = ei[e_r];
        const int cidx = ei[NE + e_r];
        segp[mi][0] = ea + (size_t)e_r * D;
        segp[mi][1] = x + (size_t)ridx * D;
        segp[mi][2] = x + (size_t)cidx * D;
    }

    const v8f vzero = {0.f, 0.f, 0.f, 0.f, 0.f, 0.f, 0.f, 0.f};
    v8f acc[MT][8];
    #pragma unroll
    for (int mi = 0; mi < MT; mi++)
        #pragma unroll
        for (int n = 0; n < 8; n++) acc[mi][n] = vzero;

    // ---- GEMM1: [64 x 384] x [384 x 128], B reused across 4 M-tiles ----
    #pragma unroll
    for (int seg = 0; seg < 3; seg++) {
        for (int ks = 0; ks < 4; ks++) {
            const int kb = seg * 128 + ks * 32;
            v16bf a[MT];
            #pragma unroll
            for (int mi = 0; mi < MT; mi++)
                a[mi] = cvt_fragA_f32(segp[mi][seg] + ks * 32 + kA, 1.0f);
            #pragma unroll
            for (int n = 0; n < 8; n++) {
                v16bf b = load_fragB_bf16(wte1, 3 * D, n * 16 + r, kb + kB);
                #pragma unroll
                for (int mi = 0; mi < MT; mi++)
                    acc[mi][n] = WMMA_BF16(a[mi], b, acc[mi][n]);
            }
        }
    }

    // ---- bias + silu -> LDS h slab (retires acc registers) ----
    __bf16* h = sh_h[wave];
    #pragma unroll
    for (int mi = 0; mi < MT; mi++)
        #pragma unroll
        for (int n = 0; n < 8; n++) {
            const float bias = sbias[n * 16 + r];
            #pragma unroll
            for (int i = 0; i < 8; i++)
                h[(mi * 16 + kA + i) * 136 + n * 16 + r] = (__bf16)silu_f(acc[mi][n][i] + bias);
        }

    // ---- GEMM2: [64 x 128] x [128 x 128], B reused across 4 M-tiles ----
    v8f acc2[MT][8];
    #pragma unroll
    for (int mi = 0; mi < MT; mi++)
        #pragma unroll
        for (int n = 0; n < 8; n++) acc2[mi][n] = vzero;
    for (int ks = 0; ks < 4; ks++) {
        v16bf a2[MT];
        #pragma unroll
        for (int mi = 0; mi < MT; mi++)
            a2[mi] = load_fragA_bf16(h + (mi * 16 + r) * 136, ks * 32 + kA);
        #pragma unroll
        for (int n = 0; n < 8; n++) {
            v16bf b = load_fragB_bf16(wte2, D, n * 16 + r, ks * 32 + kB);
            #pragma unroll
            for (int mi = 0; mi < MT; mi++)
                acc2[mi][n] = WMMA_BF16(a2[mi], b, acc2[mi][n]);
        }
    }

    // ---- per-M-tile: bias+silu -> f32 LDS staging; coalesced epilogue ----
    float* o = sh_o[wave];
    #pragma unroll
    for (int mi = 0; mi < MT; mi++) {
        #pragma unroll
        for (int n = 0; n < 8; n++) {
            const float bias = sbias[D + n * 16 + r];
            #pragma unroll
            for (int i = 0; i < 8; i++)
                o[(kA + i) * 132 + n * 16 + r] = silu_f(acc2[mi][n][i] + bias);
        }
        for (int m = 0; m < 16; m++) {
            const int e = (tile0 + mi) * 16 + m;
            const int ce = ei[NE + e];
            const float4 eav = *(const float4*)(ea + (size_t)e * D + lane * 4);
            const float4 hv = *(const float4*)(o + m * 132 + lane * 4);
            float4 v = make_float4(hv.x + eav.x, hv.y + eav.y, hv.z + eav.z, hv.w + eav.w);
            *(float4*)(out_e + (size_t)e * D + lane * 4) = v;
            float* sp = sums + (size_t)ce * D + lane * 4;
            atomicAdd(sp + 0, v.x);
            atomicAdd(sp + 1, v.y);
            atomicAdd(sp + 2, v.z);
            atomicAdd(sp + 3, v.w);
            if (lane == 0) atomicAdd(cnt + ce, 1.0f);
        }
    }
}

// ---------------------------------------------------------------------------
// Kernel 2: node MLP. One wave = one 16-node tile.
// ---------------------------------------------------------------------------
__global__ __launch_bounds__(32 * NWV) void node_kernel(
    const float* __restrict__ x, const float* __restrict__ sums, const float* __restrict__ cnt,
    const __bf16* __restrict__ wtn1, const float* __restrict__ bn1,
    const __bf16* __restrict__ wtn2, const float* __restrict__ bn2,
    float* __restrict__ out_x) {
    __shared__ __bf16 sh_h[NWV][16 * 136];
    __shared__ float  sh_o[NWV][16 * 132];
    const int wave = threadIdx.x >> 5;
    const int lane = threadIdx.x & 31;
    const int tile = blockIdx.x * NWV + wave;
    if (tile >= NN / 16) return;                      // whole-wave early exit
    const int r = lane & 15;
    const int hf = lane >> 4;
    const int kA = hf * 8;
    const int kB = hf * 8;
    const int node = tile * 16 + r;
    const float inv = 1.0f / fmaxf(cnt[node], 1.0f);  // scatter-mean divisor
    const float* segA = sums + (size_t)node * D;
    const float* segX = x + (size_t)node * D;

    const v8f vzero = {0.f, 0.f, 0.f, 0.f, 0.f, 0.f, 0.f, 0.f};
    v8f acc[8];
    #pragma unroll
    for (int n = 0; n < 8; n++) acc[n] = vzero;

    #pragma unroll
    for (int seg = 0; seg < 2; seg++) {
        const float* base = seg ? segX : segA;
        const float scale = seg ? 1.0f : inv;
        for (int ks = 0; ks < 4; ks++) {
            const int kb = seg * 128 + ks * 32;
            v16bf a = cvt_fragA_f32(base + ks * 32 + kA, scale);
            #pragma unroll
            for (int n = 0; n < 8; n++) {
                v16bf b = load_fragB_bf16(wtn1, 2 * D, n * 16 + r, kb + kB);
                acc[n] = WMMA_BF16(a, b, acc[n]);
            }
        }
    }

    __bf16* h = sh_h[wave];
    #pragma unroll
    for (int n = 0; n < 8; n++) {
        const float bias = bn1[n * 16 + r];
        #pragma unroll
        for (int i = 0; i < 8; i++)
            h[(kA + i) * 136 + n * 16 + r] = (__bf16)silu_f(acc[n][i] + bias);
    }

    v8f acc2[8];
    #pragma unroll
    for (int n = 0; n < 8; n++) acc2[n] = vzero;
    for (int ks = 0; ks < 4; ks++) {
        v16bf a = load_fragA_bf16(h + r * 136, ks * 32 + kA);
        #pragma unroll
        for (int n = 0; n < 8; n++) {
            v16bf b = load_fragB_bf16(wtn2, D, n * 16 + r, ks * 32 + kB);
            acc2[n] = WMMA_BF16(a, b, acc2[n]);
        }
    }

    float* o = sh_o[wave];
    #pragma unroll
    for (int n = 0; n < 8; n++) {
        const float bias = bn2[n * 16 + r];
        #pragma unroll
        for (int i = 0; i < 8; i++)
            o[(kA + i) * 132 + n * 16 + r] = silu_f(acc2[n][i] + bias);
    }

    for (int m = 0; m < 16; m++) {
        const int nd = tile * 16 + m;
        const float4 xv = *(const float4*)(x + (size_t)nd * D + lane * 4);
        const float4 gv = *(const float4*)(o + m * 132 + lane * 4);
        float4 v = make_float4(gv.x + xv.x, gv.y + xv.y, gv.z + xv.z, gv.w + xv.w);
        *(float4*)(out_x + (size_t)nd * D + lane * 4) = v;
    }
}

// ---------------------------------------------------------------------------
extern "C" void kernel_launch(void* const* d_in, const int* in_sizes, int n_in,
                              void* d_out, int out_size, void* d_ws, size_t ws_size,
                              hipStream_t stream) {
    const float* x   = (const float*)d_in[0];
    const int*   ei  = (const int*)d_in[1];
    const float* ea  = (const float*)d_in[2];
    const float* We1 = (const float*)d_in[3];
    const float* be1 = (const float*)d_in[4];
    const float* We2 = (const float*)d_in[5];
    const float* be2 = (const float*)d_in[6];
    const float* Wn1 = (const float*)d_in[7];
    const float* bn1 = (const float*)d_in[8];
    const float* Wn2 = (const float*)d_in[9];
    const float* bn2 = (const float*)d_in[10];

    float* out_x = (float*)d_out;                    // x_out first in return order
    float* out_e = out_x + (size_t)NN * D;           // then edge_out

    char* ws = (char*)d_ws;
    float* sums = (float*)ws;  ws += (size_t)NN * D * sizeof(float);   // 25.6 MB (L2-resident)
    float* cnt  = (float*)ws;  ws += (size_t)NN * sizeof(float);       // 200 KB
    __bf16* wte1 = (__bf16*)ws; ws += (size_t)3 * D * D * sizeof(__bf16);
    __bf16* wte2 = (__bf16*)ws; ws += (size_t)D * D * sizeof(__bf16);
    __bf16* wtn1 = (__bf16*)ws; ws += (size_t)2 * D * D * sizeof(__bf16);
    __bf16* wtn2 = (__bf16*)ws; ws += (size_t)D * D * sizeof(__bf16);

    init_kernel<<<2048, 256, 0, stream>>>(We1, We2, Wn1, Wn2, sums, cnt,
                                          wte1, wte2, wtn1, wtn2);
    edge_kernel<<<NE / (16 * MT * WV), 32 * WV, 0, stream>>>(x, ei, ea, wte1, be1, wte2, be2,
                                                             out_e, sums, cnt);
    node_kernel<<<(NN / 16 + NWV - 1) / NWV, 32 * NWV, 0, stream>>>(x, sums, cnt,
                                                                    wtn1, bn1, wtn2, bn2, out_x);
}
